// cross_attention_53352083751516
// MI455X (gfx1250) — matrix operand order repaired and checked by reading, not compiled, not run on hardware
//
#include <hip/hip_runtime.h>

// ---------------------------------------------------------------------------
// CDNA5 (gfx1250) cross-attention. All GEMMs via v_wmma_f32_16x16x32_bf16.
// wave32. GEMM waves own software-pipelined 16x64 tiles. Flash attention is
// workgroup-cooperative: 8 waves share (b,h); K/V 64-key tiles are staged to
// LDS with global_load_async_to_lds_b128 (double-buffered, ASYNCcnt).
// ---------------------------------------------------------------------------

typedef __attribute__((ext_vector_type(16))) __bf16 v16bf;
typedef __attribute__((ext_vector_type(8)))  __bf16 v8bf;
typedef __attribute__((ext_vector_type(8)))  float  v8f;
typedef __attribute__((ext_vector_type(4)))  float  v4f;

#define DIM    512
#define HEADS  8
#define DHEAD  64
#define SEQ    2048
#define BATCH  8
#define ROWS   (BATCH * SEQ)              /* 16384 */
#define SCALE  0.044194173824159216f      /* DIM^-0.5 */
#define KSTR   72                         /* padded LDS row (bank-conflict-free) */

__device__ __forceinline__ v8f wmma_bf16(v16bf a, v16bf b, v8f c) {
  return __builtin_amdgcn_wmma_f32_16x16x32_bf16(false, a, false, b,
                                                 (short)0, c, false, false);
}

// xor-lane shuffle via ds_swizzle; MASK in {1,2,4,8} stays within 16-lane half
template <int MASK>
__device__ __forceinline__ float swz_xor(float v) {
  return __int_as_float(
      __builtin_amdgcn_ds_swizzle(__float_as_int(v), (MASK << 10) | 0x1f));
}

__device__ __forceinline__ v16bf cat8(v8bf lo, v8bf hi) {
  v16bf r;
#pragma unroll
  for (int i = 0; i < 8; ++i) { r[i] = lo[i]; r[i + 8] = hi[i]; }
  return r;
}

// CDNA5 async DMA: global -> LDS, 16B per lane, tracked by ASYNCcnt.
__device__ __forceinline__ void async_b128(unsigned lds_off, const void* gaddr) {
  asm volatile("global_load_async_to_lds_b128 %0, %1, off"
               :: "v"(lds_off), "v"(gaddr) : "memory");
}
#define S_WAIT_ASYNCCNT(n) asm volatile("s_wait_asynccnt " #n ::: "memory")

// ---------------------------------------------------------------------------
// Kernel 1a: weights fp32 -> bf16 transposed (+ scale fold): WT[c][r]=W[r][c]*s
// ---------------------------------------------------------------------------
__global__ void wt_conv_kernel(const float* __restrict__ W,
                               __bf16* __restrict__ WT, float scale) {
  int t = blockIdx.x * blockDim.x + threadIdx.x;   // 0 .. 512*512-1
  int r = t >> 9, c = t & 511;
  WT[(size_t)c * DIM + r] = (__bf16)(W[(size_t)r * DIM + c] * scale);
}

// ---------------------------------------------------------------------------
// Kernel 1b: activations fp32 -> bf16 (8 elems / thread)
// ---------------------------------------------------------------------------
__global__ void act_conv_kernel(const float* __restrict__ src,
                                __bf16* __restrict__ dst) {
  size_t t = (size_t)blockIdx.x * blockDim.x + threadIdx.x;
  const float* p = src + t * 8;
  v4f a = *(const v4f*)p;
  v4f b = *(const v4f*)(p + 4);
  v8bf o;
#pragma unroll
  for (int i = 0; i < 4; ++i) { o[i] = (__bf16)a[i]; o[i + 4] = (__bf16)b[i]; }
  *(v8bf*)(dst + t * 8) = o;
}

// ---------------------------------------------------------------------------
// Kernel 2: projection GEMM (bf16 in, bf16 out), 16x64 tile per wave,
// software-pipelined k-loop (next fragments prefetched before current WMMAs).
// vmode==0: store [B,H,N,64] (Q,K)   vmode==1: store [B,H,64,J] (V^T)
// ---------------------------------------------------------------------------
__global__ void __launch_bounds__(256)
proj_kernel(const __bf16* __restrict__ src, const __bf16* __restrict__ WT,
            __bf16* __restrict__ dst, int vmode) {
  const int lane = threadIdx.x & 31;
  const int w    = blockIdx.x * 8 + (threadIdx.x >> 5);   // 8192 waves
  const int m0   = (w >> 3) << 4;
  const int n0   = (w & 7) << 6;

  const int am  = lane & 15;
  const int akb = (lane & 16) ? 16 : 0;
  const int bkb = (lane & 16) ? 16 : 0;

  const __bf16* ap = src + (size_t)(m0 + am) * DIM;
  const __bf16* bp = WT + (size_t)(n0 + (lane & 15)) * DIM + bkb;

  v8f acc[4] = {{}, {}, {}, {}};
  v16bf a = cat8(*(const v8bf*)(ap + akb), *(const v8bf*)(ap + 8 + akb));
  v16bf b0 = *(const v16bf*)(bp);
  v16bf b1 = *(const v16bf*)(bp + (size_t)16 * DIM);
  v16bf b2 = *(const v16bf*)(bp + (size_t)32 * DIM);
  v16bf b3 = *(const v16bf*)(bp + (size_t)48 * DIM);

#pragma unroll 1
  for (int k0 = 0; k0 < DIM - 32; k0 += 32) {
    const int kn = k0 + 32;
    v16bf an = cat8(*(const v8bf*)(ap + kn + akb),
                    *(const v8bf*)(ap + kn + 8 + akb));
    v16bf n0f = *(const v16bf*)(bp + kn);
    v16bf n1f = *(const v16bf*)(bp + (size_t)16 * DIM + kn);
    v16bf n2f = *(const v16bf*)(bp + (size_t)32 * DIM + kn);
    v16bf n3f = *(const v16bf*)(bp + (size_t)48 * DIM + kn);
    acc[0] = wmma_bf16(a, b0, acc[0]);
    acc[1] = wmma_bf16(a, b1, acc[1]);
    acc[2] = wmma_bf16(a, b2, acc[2]);
    acc[3] = wmma_bf16(a, b3, acc[3]);
    a = an; b0 = n0f; b1 = n1f; b2 = n2f; b3 = n3f;
  }
  acc[0] = wmma_bf16(a, b0, acc[0]);
  acc[1] = wmma_bf16(a, b1, acc[1]);
  acc[2] = wmma_bf16(a, b2, acc[2]);
  acc[3] = wmma_bf16(a, b3, acc[3]);

  const int mb = (lane & 16) ? 8 : 0;
#pragma unroll
  for (int c = 0; c < 4; ++c) {
    int bcol = n0 + 16 * c + (lane & 15);
    int h = bcol >> 6, d = bcol & 63;
#pragma unroll
    for (int r = 0; r < 8; ++r) {
      int R = m0 + mb + r;
      int bidx = R >> 11, n = R & (SEQ - 1);
      size_t o = vmode
        ? ((size_t)(bidx * HEADS + h) * DHEAD + d) * SEQ + n      // V^T
        : ((size_t)(bidx * HEADS + h) * SEQ + n) * DHEAD + d;     // Q/K
      dst[o] = (__bf16)acc[c][r];
    }
  }
}

// ---------------------------------------------------------------------------
// Kernel 3: cooperative flash attention.
// Block = 8 waves = 8 query tiles of one (b,h). Per 64-key block, K (8KB) and
// V^T (8KB) are async-DMA'd into double-buffered LDS (rows padded to 72 elems
// for conflict-free fragment reads), then each wave runs 16 WMMAs + online
// softmax with ds_swizzle reductions. SCALE pre-folded into Wq.
// ---------------------------------------------------------------------------
__global__ void __launch_bounds__(256)
attn_kernel(const __bf16* __restrict__ Q, const __bf16* __restrict__ K,
            const __bf16* __restrict__ Vt, __bf16* __restrict__ O) {
  __shared__ __attribute__((aligned(16))) __bf16 ldsK[2][64 * KSTR];
  __shared__ __attribute__((aligned(16))) __bf16 ldsV[2][64 * KSTR];
  __shared__ __attribute__((aligned(16))) __bf16 ldsP[8][16 * KSTR];

  const int tid  = threadIdx.x;
  const int lane = tid & 31;
  const int wid  = tid >> 5;
  const int blk  = blockIdx.x;                 // 1024 blocks
  const int qg = blk & 15;
  const int h  = (blk >> 4) & 7;
  const int b  = blk >> 7;
  const int n0 = (qg * 8 + wid) * 16;          // this wave's 16 query rows

  const size_t qkbase = (size_t)(b * HEADS + h) * SEQ * DHEAD;
  const size_t vbase  = (size_t)(b * HEADS + h) * DHEAD * SEQ;

  const int am  = lane & 15;
  const int akb = (lane & 16) ? 16 : 0;
  const int bkb = (lane & 16) ? 16 : 0;

  // resident Q fragments (d 0..31, 32..63); SCALE already folded into Wq
  const __bf16* qp = Q + qkbase + (size_t)(n0 + am) * DHEAD;
  v16bf qa0 = cat8(*(const v8bf*)(qp + akb),      *(const v8bf*)(qp + 8 + akb));
  v16bf qa1 = cat8(*(const v8bf*)(qp + 32 + akb), *(const v8bf*)(qp + 40 + akb));

  // async staging geometry: thread copies 32B of K and 32B of V per block
  const char* kg = (const char*)(K + qkbase);
  const char* vg = (const char*)(Vt + vbase);
  const unsigned lk0 = (unsigned)(uintptr_t)&ldsK[0][0];
  const unsigned lk1 = (unsigned)(uintptr_t)&ldsK[1][0];
  const unsigned lv0 = (unsigned)(uintptr_t)&ldsV[0][0];
  const unsigned lv1 = (unsigned)(uintptr_t)&ldsV[1][0];
  const int rsel = tid >> 2;            // row (key for K, d for V)
  const int csel = (tid & 3) * 32;      // byte offset within 128B row

  auto stage = [&](int j0, int buf) {
    const char* gk = kg + (size_t)j0 * (DHEAD * 2) + (size_t)tid * 32;
    unsigned lk = (buf ? lk1 : lk0) + rsel * (KSTR * 2) + csel;
    async_b128(lk, gk);
    async_b128(lk + 16, gk + 16);
    const char* gv = vg + (size_t)rsel * (SEQ * 2) + (size_t)j0 * 2 + csel;
    unsigned lv = (buf ? lv1 : lv0) + rsel * (KSTR * 2) + csel;
    async_b128(lv, gv);
    async_b128(lv + 16, gv + 16);
  };

  v8f acc[4] = {{}, {}, {}, {}};
  float mst[8], lst[8];
#pragma unroll
  for (int r = 0; r < 8; ++r) { mst[r] = -1e30f; lst[r] = 0.f; }

  __bf16* pl = &ldsP[wid][0];

  stage(0, 0);
  int cur = 0;
  for (int j0 = 0; j0 < SEQ; j0 += 64, cur ^= 1) {
    if (j0 + 64 < SEQ) {        // prefetch next block into other buffer
      stage(j0 + 64, cur ^ 1);
      S_WAIT_ASYNCCNT(4);       // 4 newest (next-block) may remain in flight
    } else {
      S_WAIT_ASYNCCNT(0);
    }
    __syncthreads();            // LDS tiles of `cur` valid for all waves

    const __bf16* kt = ldsK[cur];
    const __bf16* vt = ldsV[cur];

    // ---- QK^T: four 16-key tiles, K-dim 64 split across two WMMAs ----
    v8f d[4];
    const v8f z = {};
#pragma unroll
    for (int t = 0; t < 4; ++t) {
      const __bf16* kr = kt + (16 * t + am) * KSTR;
      d[t] = wmma_bf16(qa0, cat8(*(const v8bf*)(kr + bkb),
                                 *(const v8bf*)(kr + bkb + 8)), z);
      d[t] = wmma_bf16(qa1, cat8(*(const v8bf*)(kr + 32 + bkb),
                                 *(const v8bf*)(kr + 40 + bkb)), d[t]);
    }

    // ---- online softmax over 64 keys ----
    float p[4][8];
    v8f corrv = {};
#pragma unroll
    for (int r = 0; r < 8; ++r) {
      float x = fmaxf(fmaxf(d[0][r], d[1][r]), fmaxf(d[2][r], d[3][r]));
      x = fmaxf(x, swz_xor<1>(x));
      x = fmaxf(x, swz_xor<2>(x));
      x = fmaxf(x, swz_xor<4>(x));
      x = fmaxf(x, swz_xor<8>(x));
      float mnew = fmaxf(mst[r], x);
      float corr = __expf(mst[r] - mnew);
      float s = 0.f;
#pragma unroll
      for (int t = 0; t < 4; ++t) { p[t][r] = __expf(d[t][r] - mnew); s += p[t][r]; }
      s += swz_xor<1>(s);
      s += swz_xor<2>(s);
      s += swz_xor<4>(s);
      s += swz_xor<8>(s);
      lst[r] = lst[r] * corr + s;
      mst[r] = mnew;
      corrv[r] = corr;
    }
#pragma unroll
    for (int c = 0; c < 4; ++c) acc[c] *= corrv;

    // ---- P: C-layout -> LDS (16 x 64, padded rows) -> two A-fragments ----
    const int mb = (lane & 16) ? 8 : 0;
    const int cb = lane & 15;
#pragma unroll
    for (int t = 0; t < 4; ++t)
#pragma unroll
      for (int r = 0; r < 8; ++r)
        pl[(mb + r) * KSTR + 16 * t + cb] = (__bf16)p[t][r];

    v16bf pa0 = cat8(*(const v8bf*)&pl[am * KSTR + akb],
                     *(const v8bf*)&pl[am * KSTR + 8 + akb]);
    v16bf pa1 = cat8(*(const v8bf*)&pl[am * KSTR + 32 + akb],
                     *(const v8bf*)&pl[am * KSTR + 40 + akb]);

    // ---- P @ V from LDS: four 16-wide d tiles x two 32-key chunks ----
#pragma unroll
    for (int c = 0; c < 4; ++c) {
      const __bf16* vr = vt + (16 * c + am) * KSTR;
      acc[c] = wmma_bf16(pa0, cat8(*(const v8bf*)(vr + bkb),
                                   *(const v8bf*)(vr + bkb + 8)), acc[c]);
      acc[c] = wmma_bf16(pa1, cat8(*(const v8bf*)(vr + 32 + bkb),
                                   *(const v8bf*)(vr + 40 + bkb)), acc[c]);
    }

    __syncthreads();            // all waves done with `cur` before restaging
  }

  // ---- epilogue: normalize, store O as [B*N, H*64] bf16 ----
  v8f invl;
#pragma unroll
  for (int r = 0; r < 8; ++r) invl[r] = 1.0f / lst[r];

  const int mb = (lane & 16) ? 8 : 0;
  const int cb = lane & 15;
#pragma unroll
  for (int c = 0; c < 4; ++c) {
    acc[c] *= invl;
#pragma unroll
    for (int r = 0; r < 8; ++r) {
      size_t row = (size_t)b * SEQ + n0 + mb + r;
      O[row * (HEADS * DHEAD) + h * DHEAD + 16 * c + cb] = (__bf16)acc[c][r];
    }
  }
}

// ---------------------------------------------------------------------------
// Kernel 4: output projection  out = O(bf16) @ Wo + bo  (fp32 out), pipelined
// ---------------------------------------------------------------------------
__global__ void __launch_bounds__(256)
oproj_kernel(const __bf16* __restrict__ Obf, const __bf16* __restrict__ WoT,
             const float* __restrict__ bo, float* __restrict__ out) {
  const int lane = threadIdx.x & 31;
  const int w    = blockIdx.x * 8 + (threadIdx.x >> 5);   // 8192 waves
  const int m0   = (w >> 3) << 4;
  const int n0   = (w & 7) << 6;

  const int am  = lane & 15;
  const int akb = (lane & 16) ? 16 : 0;
  const int bkb = (lane & 16) ? 16 : 0;

  const __bf16* ap = Obf + (size_t)(m0 + am) * DIM;
  const __bf16* bp = WoT + (size_t)(n0 + (lane & 15)) * DIM + bkb;

  v8f acc[4] = {{}, {}, {}, {}};
  v16bf a = cat8(*(const v8bf*)(ap + akb), *(const v8bf*)(ap + 8 + akb));
  v16bf b0 = *(const v16bf*)(bp);
  v16bf b1 = *(const v16bf*)(bp + (size_t)16 * DIM);
  v16bf b2 = *(const v16bf*)(bp + (size_t)32 * DIM);
  v16bf b3 = *(const v16bf*)(bp + (size_t)48 * DIM);

#pragma unroll 1
  for (int k0 = 0; k0 < DIM - 32; k0 += 32) {
    const int kn = k0 + 32;
    v16bf an = cat8(*(const v8bf*)(ap + kn + akb),
                    *(const v8bf*)(ap + kn + 8 + akb));
    v16bf n0f = *(const v16bf*)(bp + kn);
    v16bf n1f = *(const v16bf*)(bp + (size_t)16 * DIM + kn);
    v16bf n2f = *(const v16bf*)(bp + (size_t)32 * DIM + kn);
    v16bf n3f = *(const v16bf*)(bp + (size_t)48 * DIM + kn);
    acc[0] = wmma_bf16(a, b0, acc[0]);
    acc[1] = wmma_bf16(a, b1, acc[1]);
    acc[2] = wmma_bf16(a, b2, acc[2]);
    acc[3] = wmma_bf16(a, b3, acc[3]);
    a = an; b0 = n0f; b1 = n1f; b2 = n2f; b3 = n3f;
  }
  acc[0] = wmma_bf16(a, b0, acc[0]);
  acc[1] = wmma_bf16(a, b1, acc[1]);
  acc[2] = wmma_bf16(a, b2, acc[2]);
  acc[3] = wmma_bf16(a, b3, acc[3]);

  const int mb = (lane & 16) ? 8 : 0;
#pragma unroll
  for (int c = 0; c < 4; ++c) {
    int bcol = n0 + 16 * c + (lane & 15);
    float bias = bo[bcol];
#pragma unroll
    for (int r = 0; r < 8; ++r)
      out[(size_t)(m0 + mb + r) * DIM + bcol] = acc[c][r] + bias;
  }
}

// ---------------------------------------------------------------------------
// launch
// ---------------------------------------------------------------------------
extern "C" void kernel_launch(void* const* d_in, const int* in_sizes, int n_in,
                              void* d_out, int out_size, void* d_ws, size_t ws_size,
                              hipStream_t stream) {
  const float* x  = (const float*)d_in[0];
  const float* m  = (const float*)d_in[1];
  const float* Wq = (const float*)d_in[2];
  const float* Wk = (const float*)d_in[3];
  const float* Wv = (const float*)d_in[4];
  const float* Wo = (const float*)d_in[5];
  const float* bo = (const float*)d_in[6];
  float* out = (float*)d_out;

  char* ws = (char*)d_ws;
  const size_t WSZ = (size_t)DIM * DIM * 2;    // 512 KB  (bf16 weight)
  const size_t ASZ = (size_t)ROWS * DIM * 2;   // 16 MB   (bf16 activation)
  __bf16* WqT = (__bf16*)(ws);
  __bf16* WkT = (__bf16*)(ws + WSZ);
  __bf16* WvT = (__bf16*)(ws + 2 * WSZ);
  __bf16* WoT = (__bf16*)(ws + 3 * WSZ);
  __bf16* Xb  = (__bf16*)(ws + 4 * WSZ);
  __bf16* Mb  = (__bf16*)(ws + 4 * WSZ + 1 * ASZ);
  __bf16* Qb  = (__bf16*)(ws + 4 * WSZ + 2 * ASZ);
  __bf16* Kb  = (__bf16*)(ws + 4 * WSZ + 3 * ASZ);
  __bf16* Vtb = (__bf16*)(ws + 4 * WSZ + 4 * ASZ);
  __bf16* Ob  = (__bf16*)(ws + 4 * WSZ + 5 * ASZ);

  dim3 blk(256);  // 8 wave32 waves per workgroup

  wt_conv_kernel<<<1024, blk, 0, stream>>>(Wq, WqT, SCALE);  // scale folded
  wt_conv_kernel<<<1024, blk, 0, stream>>>(Wk, WkT, 1.0f);
  wt_conv_kernel<<<1024, blk, 0, stream>>>(Wv, WvT, 1.0f);
  wt_conv_kernel<<<1024, blk, 0, stream>>>(Wo, WoT, 1.0f);

  act_conv_kernel<<<4096, blk, 0, stream>>>(x, Xb);
  act_conv_kernel<<<4096, blk, 0, stream>>>(m, Mb);

  proj_kernel<<<1024, blk, 0, stream>>>(Xb, WqT, Qb, 0);   // Q [B,H,N,64]
  proj_kernel<<<1024, blk, 0, stream>>>(Mb, WkT, Kb, 0);   // K [B,H,J,64]
  proj_kernel<<<1024, blk, 0, stream>>>(Mb, WvT, Vtb, 1);  // V^T [B,H,64,J]

  attn_kernel<<<1024, blk, 0, stream>>>(Qb, Kb, Vtb, Ob);

  oproj_kernel<<<1024, blk, 0, stream>>>(Ob, WoT, bo, out);

  (void)in_sizes; (void)n_in; (void)out_size; (void)ws_size;
}
